// MultiheadSelfAttention_10634339025525
// MI455X (gfx1250) — matrix operand order repaired and checked by reading, not compiled, not run on hardware
//
#include <hip/hip_runtime.h>

typedef _Float16 half_t;
typedef __attribute__((ext_vector_type(16))) _Float16 v16h;
typedef __attribute__((ext_vector_type(8)))  _Float16 v8h;
typedef __attribute__((ext_vector_type(8)))  float    v8f;

#define BB 4
#define SS 2048
#define DD 1024
#define HH 16
#define DH 64

__device__ __forceinline__ v8f wmma32(v16h a, v16h b, v8f c) {
  // D = A(16x32 f16) * B(32x16 f16) + C(16x16 f32)
  return __builtin_amdgcn_wmma_f32_16x16x32_f16(false, a, false, b, (short)0, c,
                                                false, false);
}

// A-fragment / LDS-P-fragment: two 8-half runs at +0 and +16 (halves)
__device__ __forceinline__ v16h frag_pair(const half_t* p) {
  v8h lo = *(const v8h*)(p);
  v8h hi = *(const v8h*)(p + 16);
  v16h r;
#pragma unroll
  for (int i = 0; i < 8; ++i) { r[i] = lo[i]; r[i + 8] = hi[i]; }
  return r;
}

// B-fragment: two 8-half runs at +0 and +16 (halves)
__device__ __forceinline__ v16h frag_contig(const half_t* p) {
  v8h lo = *(const v8h*)(p);
  v8h hi = *(const v8h*)(p + 16);
  v16h r;
#pragma unroll
  for (int i = 0; i < 8; ++i) { r[i] = lo[i]; r[i + 8] = hi[i]; }
  return r;
}

__global__ void cvt_f32_f16(const float* __restrict__ src, half_t* __restrict__ dst, int n) {
  int i = blockIdx.x * blockDim.x + threadIdx.x;
  int stride = gridDim.x * blockDim.x;
  for (; i < n; i += stride) dst[i] = (half_t)src[i];
}

// ---------------------------------------------------------------------------
// QKV projection (+RoPE on Q,K; V stored transposed Dh x S per (b,h)).
// grid = (512, 1, 3); block = 128 (4 waves). Wave tile: 64 rows x 64 cols.
// Per K=32 step: 8 A-loads + 8 B-loads (b128) feed 16 WMMAs.
// ---------------------------------------------------------------------------
__global__ __launch_bounds__(128) void qkv_proj_rope(
    const half_t* __restrict__ Xh,
    const half_t* __restrict__ Wqh, const half_t* __restrict__ Wkh,
    const half_t* __restrict__ Wvh, const int* __restrict__ tp,
    half_t* __restrict__ Qh, half_t* __restrict__ Kh, half_t* __restrict__ VTh) {
  const int lane = threadIdx.x & 31;
  const int wid  = threadIdx.x >> 5;
  const int mode = blockIdx.z;                   // 0=Q, 1=K, 2=V
  const half_t* W = (mode == 0) ? Wqh : (mode == 1) ? Wkh : Wvh;

  const int tile = blockIdx.x * 4 + wid;         // 0..2047
  const int rowt = tile >> 4;                    // 0..127 (64 rows each over B*S)
  const int head = tile & 15;
  const int r0   = rowt * 64;
  const int n0   = lane & 15;
  const int bk   = (lane >> 4) << 3;             // A frag k base: 0/8
  const int bo   = (lane >> 4) << 3;             // B frag k base: 0/8
  const int mb   = bk;                           // C/D row base: 0/8

  v8f acc[4][4] = {};
  const half_t* arow = Xh + (size_t)(r0 + n0) * DD;
  const half_t* wrow = W + (size_t)(head * 64 + n0) * DD;

#pragma unroll 1
  for (int dk = 0; dk < DD; dk += 32) {
    v16h a[4], bf[4];
#pragma unroll
    for (int mt = 0; mt < 4; ++mt)
      a[mt] = frag_pair(arow + (size_t)mt * 16 * DD + dk + bk);
#pragma unroll
    for (int c = 0; c < 4; ++c)
      bf[c] = frag_contig(wrow + (size_t)c * 16 * DD + dk + bo);
#pragma unroll
    for (int mt = 0; mt < 4; ++mt)
#pragma unroll
      for (int c = 0; c < 4; ++c)
        acc[mt][c] = wmma32(a[mt], bf[c], acc[mt][c]);
  }

  const int b  = r0 / SS;
  const int s0 = r0 % SS;
  const int bh = b * HH + head;

  if (mode < 2) {
    half_t* dst = ((mode == 0) ? Qh : Kh) + (size_t)bh * SS * DH;
#pragma unroll
    for (int c = 0; c < 4; ++c) {
      const int j = c * 16 + n0;
      const float freq = __expf((float)(j & ~1) * -0.14391156515f); // theta^{-2i/64}
#pragma unroll
      for (int mt = 0; mt < 4; ++mt) {
#pragma unroll
        for (int v = 0; v < 8; ++v) {
          const int s = s0 + mt * 16 + mb + v;
          float x  = acc[mt][c][v];
          float xo = __shfl_xor(x, 1);           // partner of the RoPE pair
          float sn, cs;
          __sincosf((float)tp[s] * freq, &sn, &cs);
          float r = (j & 1) ? (xo * sn + x * cs) // odd slot:  xe*sin + xo*cos
                            : (x * cs - xo * sn);// even slot: xe*cos - xo*sin
          dst[(size_t)s * DH + j] = (half_t)r;
        }
      }
    }
  } else {
    // V^T: (b,h,d,s) — per-lane 8 consecutive s values -> one b128 store
#pragma unroll
    for (int c = 0; c < 4; ++c) {
      const int j = c * 16 + n0;
#pragma unroll
      for (int mt = 0; mt < 4; ++mt) {
        v8h pk;
#pragma unroll
        for (int v = 0; v < 8; ++v) pk[v] = (half_t)acc[mt][c][v];
        *(v8h*)(VTh + ((size_t)bh * DH + j) * SS + s0 + mt * 16 + mb) = pk;
      }
    }
  }
}

// ---------------------------------------------------------------------------
// Flash attention, causal. One wave per (b,h,32-row q block); 32 keys/step.
// grid = 1024; block = 128 (4 waves). K/V fragments shared across 2 q-tiles:
// per step 16 b128 loads feed 16 WMMAs. O written as (b,s,h*64+d) f16.
// ---------------------------------------------------------------------------
__global__ __launch_bounds__(128) void flash_attn(
    const half_t* __restrict__ Qh, const half_t* __restrict__ Kh,
    const half_t* __restrict__ VTh, half_t* __restrict__ Oh) {
  __shared__ _Float16 Pl[4][32 * 40];            // per-wave 32x32 P tile, stride 40
  const int lane = threadIdx.x & 31;
  const int wid  = threadIdx.x >> 5;
  const int gt = blockIdx.x * 4 + wid;           // 0..4095
  const int bh = gt >> 6;                        // 0..63
  const int q0 = (gt & 63) * 32;
  const int n0 = lane & 15;
  const int bk = (lane >> 4) << 3;
  const int bo = (lane >> 4) << 3;
  const int mb = bk;

  // Q A-fragments for 2 q-tiles x {d0..31, d32..63}, pre-scaled by 1/sqrt(64)
  v16h aq[2][2];
  const _Float16 sc = (_Float16)0.125f;
#pragma unroll
  for (int qi = 0; qi < 2; ++qi) {
    const half_t* qrow = Qh + ((size_t)bh * SS + q0 + qi * 16 + n0) * DH;
    aq[qi][0] = frag_pair(qrow + bk);
    aq[qi][1] = frag_pair(qrow + 32 + bk);
#pragma unroll
    for (int i = 0; i < 16; ++i) { aq[qi][0][i] *= sc; aq[qi][1][i] *= sc; }
  }

  v8f o[2][4] = {};
  float rowmax[2][8], rowsum[2][8];
#pragma unroll
  for (int qi = 0; qi < 2; ++qi)
#pragma unroll
    for (int v = 0; v < 8; ++v) { rowmax[qi][v] = -3.0e38f; rowsum[qi][v] = 0.f; }

  const half_t* kb = Kh + (size_t)bh * SS * DH;
  const half_t* vb = VTh + (size_t)bh * DH * SS;
  half_t* pl = &Pl[wid][0];

  for (int k0 = 0; k0 <= q0 + 31; k0 += 32) {
    // K B-fragments: kf[t*2+d]: key tile t (cols k0+16t..), d-chunk d (32 each)
    const half_t* kr0 = kb + (size_t)(k0 + n0) * DH + bo;
    const half_t* kr1 = kb + (size_t)(k0 + 16 + n0) * DH + bo;
    v16h kf0a = frag_contig(kr0),      kf0b = frag_contig(kr0 + 32);
    v16h kf1a = frag_contig(kr1),      kf1b = frag_contig(kr1 + 32);

    v8f st[2][2] = {};
#pragma unroll
    for (int qi = 0; qi < 2; ++qi) {
      st[qi][0] = wmma32(aq[qi][0], kf0a, st[qi][0]);
      st[qi][0] = wmma32(aq[qi][1], kf0b, st[qi][0]);
      st[qi][1] = wmma32(aq[qi][0], kf1a, st[qi][1]);
      st[qi][1] = wmma32(aq[qi][1], kf1b, st[qi][1]);
    }

    // online softmax over the 32 columns; P written straight into LDS tile
#pragma unroll
    for (int qi = 0; qi < 2; ++qi) {
#pragma unroll
      for (int v = 0; v < 8; ++v) {
        const int m = qi * 16 + mb + v;
        const int q = q0 + m;
        float sa = st[qi][0][v], sb = st[qi][1][v];
        if (k0 + n0 > q)      sa = -1.0e30f;     // causal mask
        if (k0 + 16 + n0 > q) sb = -1.0e30f;
        float tm = fmaxf(sa, sb);
#pragma unroll
        for (int off = 1; off < 16; off <<= 1) tm = fmaxf(tm, __shfl_xor(tm, off));
        float nm   = fmaxf(rowmax[qi][v], tm);
        float corr = __expf(rowmax[qi][v] - nm);
        rowmax[qi][v] = nm;
        float pa = __expf(sa - nm);
        float pb = __expf(sb - nm);
        float ps = pa + pb;
#pragma unroll
        for (int off = 1; off < 16; off <<= 1) ps += __shfl_xor(ps, off);
        rowsum[qi][v] = rowsum[qi][v] * corr + ps;
        o[qi][0][v] *= corr; o[qi][1][v] *= corr;
        o[qi][2][v] *= corr; o[qi][3][v] *= corr;
        pl[m * 40 + n0]      = (half_t)pa;       // D-layout -> LDS (transpose)
        pl[m * 40 + 16 + n0] = (half_t)pb;
      }
    }

    // P A-fragments from LDS (same-wave DS ops are in-order; no barrier)
    v16h ap0 = frag_pair(pl + (0 * 16 + n0) * 40 + bk);
    v16h ap1 = frag_pair(pl + (1 * 16 + n0) * 40 + bk);

    // V B-fragments from V^T rows (contiguous k), shared by both q-tiles
    v16h vf[4];
#pragma unroll
    for (int c = 0; c < 4; ++c)
      vf[c] = frag_contig(vb + (size_t)(c * 16 + n0) * SS + k0 + bo);
#pragma unroll
    for (int c = 0; c < 4; ++c) {
      o[0][c] = wmma32(ap0, vf[c], o[0][c]);
      o[1][c] = wmma32(ap1, vf[c], o[1][c]);
    }
  }

  const int b = bh >> 4, h = bh & 15;
  half_t* ob = Oh + ((size_t)b * SS + q0) * DD + h * DH;
#pragma unroll
  for (int qi = 0; qi < 2; ++qi)
#pragma unroll
    for (int v = 0; v < 8; ++v) {
      const float inv = 1.0f / rowsum[qi][v];
      const size_t row = (size_t)(qi * 16 + mb + v) * DD;
#pragma unroll
      for (int c = 0; c < 4; ++c)
        ob[row + c * 16 + n0] = (half_t)(o[qi][c][v] * inv);
    }
}

// ---------------------------------------------------------------------------
// Output projection: out[r, d] = sum_e Oh[r, e] * Wo[d, e], fp32 result.
// Wave tile 64x64, same structure as qkv_proj_rope. grid = 512.
// ---------------------------------------------------------------------------
__global__ __launch_bounds__(128) void out_proj(
    const half_t* __restrict__ Ohalf, const half_t* __restrict__ Woh,
    float* __restrict__ out) {
  const int lane = threadIdx.x & 31;
  const int wid  = threadIdx.x >> 5;
  const int tile = blockIdx.x * 4 + wid;         // 0..2047
  const int rowt = tile >> 4;
  const int ct   = tile & 15;                    // 64-col tile
  const int r0   = rowt * 64;
  const int n0   = lane & 15;
  const int bk   = (lane >> 4) << 3;
  const int bo   = (lane >> 4) << 3;
  const int mb   = bk;

  v8f acc[4][4] = {};
  const half_t* arow = Ohalf + (size_t)(r0 + n0) * DD;
  const half_t* wrow = Woh + (size_t)(ct * 64 + n0) * DD;

#pragma unroll 1
  for (int dk = 0; dk < DD; dk += 32) {
    v16h a[4], bf[4];
#pragma unroll
    for (int mt = 0; mt < 4; ++mt)
      a[mt] = frag_pair(arow + (size_t)mt * 16 * DD + dk + bk);
#pragma unroll
    for (int c = 0; c < 4; ++c)
      bf[c] = frag_contig(wrow + (size_t)c * 16 * DD + dk + bo);
#pragma unroll
    for (int mt = 0; mt < 4; ++mt)
#pragma unroll
      for (int c = 0; c < 4; ++c)
        acc[mt][c] = wmma32(a[mt], bf[c], acc[mt][c]);
  }

#pragma unroll
  for (int mt = 0; mt < 4; ++mt)
#pragma unroll
    for (int c = 0; c < 4; ++c)
#pragma unroll
      for (int v = 0; v < 8; ++v)
        out[(size_t)(r0 + mt * 16 + mb + v) * DD + ct * 64 + c * 16 + n0] =
            acc[mt][c][v];
}

extern "C" void kernel_launch(void* const* d_in, const int* in_sizes, int n_in,
                              void* d_out, int out_size, void* d_ws, size_t ws_size,
                              hipStream_t stream) {
  const float* X  = (const float*)d_in[0];
  const float* Wq = (const float*)d_in[1];
  const float* Wk = (const float*)d_in[2];
  const float* Wv = (const float*)d_in[3];
  const float* Wo = (const float*)d_in[4];
  const int*   tp = (const int*)d_in[5];
  float* out = (float*)d_out;

  const size_t XSZ = (size_t)BB * SS * DD;   // 8,388,608 halves
  const size_t WSZ = (size_t)DD * DD;        // 1,048,576 halves
  half_t* ws  = (half_t*)d_ws;
  half_t* Xh  = ws;  ws += XSZ;
  half_t* Wqh = ws;  ws += WSZ;
  half_t* Wkh = ws;  ws += WSZ;
  half_t* Wvh = ws;  ws += WSZ;
  half_t* Woh = ws;  ws += WSZ;
  half_t* Qh  = ws;  ws += XSZ;
  half_t* Kh  = ws;  ws += XSZ;
  half_t* VTh = ws;  ws += XSZ;
  half_t* Oh  = ws;  ws += XSZ;

  cvt_f32_f16<<<2048, 256, 0, stream>>>(X,  Xh,  (int)XSZ);
  cvt_f32_f16<<<512,  256, 0, stream>>>(Wq, Wqh, (int)WSZ);
  cvt_f32_f16<<<512,  256, 0, stream>>>(Wk, Wkh, (int)WSZ);
  cvt_f32_f16<<<512,  256, 0, stream>>>(Wv, Wvh, (int)WSZ);
  cvt_f32_f16<<<512,  256, 0, stream>>>(Wo, Woh, (int)WSZ);

  dim3 pgrid(512, 1, 3);                     // 2048 wave-tiles per projection
  qkv_proj_rope<<<pgrid, 128, 0, stream>>>(Xh, Wqh, Wkh, Wvh, tp, Qh, Kh, VTh);
  flash_attn<<<1024, 128, 0, stream>>>(Qh, Kh, VTh, Oh);
  out_proj<<<512, 128, 0, stream>>>(Oh, Woh, out);
}
